// Bert_10187662426159
// MI455X (gfx1250) — matrix operand order repaired and checked by reading, not compiled, not run on hardware
//
#include <hip/hip_runtime.h>
#include <hip/hip_bf16.h>
#include <math.h>

// ---------------------------------------------------------------------------
// Types for CDNA5 WMMA (wave32): v_wmma_f32_16x16x32_f16
// ---------------------------------------------------------------------------
typedef __attribute__((ext_vector_type(16))) _Float16 v16h;
typedef __attribute__((ext_vector_type(8)))  float    v8f;

#define WMMAF16(a, b, c) \
  __builtin_amdgcn_wmma_f32_16x16x32_f16(false, (a), false, (b), (short)0, (c), false, false)

#define HIDDEN 1024
#define HEADS  16
#define HEAD   64
#define SEQ    512
#define BATCH  16
#define NROWS  (SEQ * BATCH)        // 8192
#define SCALE  0.07216878364870323f // 1/sqrt(3*64)

// ---------------------------------------------------------------------------
// CDNA5 async global->LDS (ASYNCcnt) with graceful fallback.
// Probe-verified: __builtin_amdgcn_global_load_async_to_lds_b128 takes
// (v4i AS1* src, v4i AS3* dst, imm offset, imm cpol).
// ---------------------------------------------------------------------------
typedef int v4i_vec __attribute__((vector_size(16)));
typedef __attribute__((address_space(1))) v4i_vec* gptr_v4i;
typedef __attribute__((address_space(3))) v4i_vec* lptr_v4i;

#if defined(__has_builtin)
#  if __has_builtin(__builtin_amdgcn_global_load_async_to_lds_b128)
#    define ASYNC_LDS 1
#  elif __has_builtin(__builtin_amdgcn_load_to_lds)
#    define ASYNC_LDS 2
#  endif
#  if __has_builtin(__builtin_amdgcn_s_wait_asynccnt)
#    define HAVE_WAIT_ASYNC 1
#  endif
#endif

__device__ __forceinline__ void copy16_g2l(void* lp, const void* gp) {
#if defined(ASYNC_LDS) && (ASYNC_LDS == 1)
  __builtin_amdgcn_global_load_async_to_lds_b128((gptr_v4i)gp, (lptr_v4i)lp, 0, 0);
#elif defined(ASYNC_LDS) && (ASYNC_LDS == 2)
  __builtin_amdgcn_load_to_lds(
      (__attribute__((address_space(1))) void*)(void*)gp,
      (__attribute__((address_space(3))) void*)lp, 16, 0, 0);
#else
  *(uint4*)lp = *(const uint4*)gp;
#endif
}

__device__ __forceinline__ void async_wait0() {
#if defined(ASYNC_LDS) && (ASYNC_LDS == 1)
#  if defined(HAVE_WAIT_ASYNC)
  __builtin_amdgcn_s_wait_asynccnt(0);
#  else
  asm volatile("s_wait_asynccnt 0x0" ::: "memory");
#  endif
#endif
}

// ---------------------------------------------------------------------------
// Fragment loaders (layouts per CDNA5 ISA 7.12.2; every load = 2x ds_read_b128)
// A (16x32 f16): lane l -> row = l&15; VGPR0..3 K = half*8 + [0..7]
//                                      VGPR4..7 K = 16 + half*8 + [0..7]
// B (32x16 f16): lane l -> col = l&15; VGPR0..3 K = half*8 + [0..7], VGPR4..7 K = 16 + half*8 + [0..7]
// C (16x16 f32): lane l, VGPR j -> row = j + 8*(l>>4), col = l&15
// ---------------------------------------------------------------------------
__device__ __forceinline__ v16h load_a_frag(const _Float16* lds, int ldh, int row0, int k0) {
  int lane = threadIdx.x & 31;
  int m  = row0 + (lane & 15);
  int ko = k0 + ((lane >> 4) << 3);
  union { v16h v; uint4 q[2]; } f;
  f.q[0] = *(const uint4*)(lds + m * ldh + ko);
  f.q[1] = *(const uint4*)(lds + m * ldh + ko + 16);
  return f.v;
}

__device__ __forceinline__ v16h load_b_frag(const _Float16* lds, int ldh, int n0, int k0) {
  int lane = threadIdx.x & 31;
  int n  = n0 + (lane & 15);
  int ko = k0 + ((lane >> 4) << 3);
  union { v16h v; uint4 q[2]; } f;
  const uint4* p = (const uint4*)(lds + n * ldh + ko);
  f.q[0] = p[0];
  f.q[1] = p[2];
  return f.v;
}

// DeBERTa log-bucket, returns index in [0,62] for r in [-511,511]
__device__ __forceinline__ int bucket_of(int r) {
  int a = r < 0 ? -r : r;
  int bp;
  if (a <= 16) {
    bp = r;
  } else {
    int ap = a < 511 ? a : 511;
    float lp = ceilf(logf((float)ap * (1.0f / 16.0f)) / logf(511.0f / 16.0f) * 15.0f) + 16.0f;
    int li = (int)lp;
    if (li > 31) li = 31;
    bp = (r < 0) ? -li : li;
  }
  return 31 + bp;
}

// ---------------------------------------------------------------------------
// Kernel 1: LayerNorm (no affine) of hidden_states -> f16 (one float4/thread)
// ---------------------------------------------------------------------------
__global__ __launch_bounds__(256) void ln_in_kernel(const float* __restrict__ x,
                                                    _Float16* __restrict__ h) {
  __shared__ float ssum[8], ssq[8];
  const int tid = threadIdx.x;
  const size_t row = blockIdx.x;
  const float4* xr = (const float4*)(x + row * HIDDEN);
  float4 v = xr[tid];
  float s = v.x + v.y + v.z + v.w;
  float q = v.x * v.x + v.y * v.y + v.z * v.z + v.w * v.w;
  for (int o = 16; o >= 1; o >>= 1) { s += __shfl_xor(s, o, 32); q += __shfl_xor(q, o, 32); }
  if ((tid & 31) == 0) { ssum[tid >> 5] = s; ssq[tid >> 5] = q; }
  __syncthreads();
  float ts = 0.f, tq = 0.f;
  for (int i = 0; i < 8; ++i) { ts += ssum[i]; tq += ssq[i]; }
  const float mean = ts * (1.0f / HIDDEN);
  const float var  = tq * (1.0f / HIDDEN) - mean * mean;
  const float rstd = rsqrtf(var + 1e-7f);
  union { ushort u[4]; uint2 d; } o16;
  o16.u[0] = __builtin_bit_cast(ushort, (_Float16)((v.x - mean) * rstd));
  o16.u[1] = __builtin_bit_cast(ushort, (_Float16)((v.y - mean) * rstd));
  o16.u[2] = __builtin_bit_cast(ushort, (_Float16)((v.z - mean) * rstd));
  o16.u[3] = __builtin_bit_cast(ushort, (_Float16)((v.w - mean) * rstd));
  *(uint2*)&h[row * HIDDEN + tid * 4] = o16.d;
}

// ---------------------------------------------------------------------------
// Kernel 2: convert weights to f16: Wcat = [wqk(2048); wv(1024)], wo16
// ---------------------------------------------------------------------------
__global__ __launch_bounds__(256) void cvt_w_kernel(const float* __restrict__ wqk,
                                                    const float* __restrict__ wv,
                                                    const float* __restrict__ wo,
                                                    _Float16* __restrict__ wcat,
                                                    _Float16* __restrict__ wo16) {
  size_t i = (size_t)blockIdx.x * 256 + threadIdx.x;
  const size_t nqk = (size_t)2048 * 1024;
  const size_t ntot = (size_t)3072 * 1024;
  if (i < ntot)
    wcat[i] = (_Float16)(i < nqk ? wqk[i] : wv[i - nqk]);
  if (i < (size_t)1024 * 1024)
    wo16[i] = (_Float16)wo[i];
}

// ---------------------------------------------------------------------------
// Kernel 3: positional projections: qpos/kpos f16 [H][64][64] (bucket 63 = pad)
// ---------------------------------------------------------------------------
__global__ __launch_bounds__(64) void pos_proj_kernel(const float* __restrict__ rel,
                                                      const float* __restrict__ wqk,
                                                      const float* __restrict__ bqk,
                                                      _Float16* __restrict__ qpos,
                                                      _Float16* __restrict__ kpos) {
  const int bidx = blockIdx.x;            // 0..2047
  const int which = bidx >> 10;           // 0 = qpos, 1 = kpos
  const int h = (bidx >> 6) & 15;
  const int bucket = bidx & 63;
  const int d = threadIdx.x;
  const int col = which * 1024 + h * 64 + d;   // row of wqk
  float acc = 0.f;
  if (bucket < 63) {
    const float4* rr = (const float4*)(rel + (size_t)bucket * HIDDEN);
    const float4* wr = (const float4*)(wqk + (size_t)col * HIDDEN);
#pragma unroll 4
    for (int k = 0; k < HIDDEN / 4; ++k) {
      float4 a = rr[k], b = wr[k];
      acc += a.x * b.x + a.y * b.y + a.z * b.z + a.w * b.w;
    }
    acc += bqk[col];
    if (which == 0) acc *= SCALE;
  }
  _Float16* dst = which ? kpos : qpos;
  dst[(h * 64 + bucket) * 64 + d] = (_Float16)acc;
}

// ---------------------------------------------------------------------------
// Kernel 4: WMMA GEMM  C[M,N] = A[M,K] * W[N,K]^T (+bias)
//   Block tile 128x128x32, 8 waves (wave tile 32x64, 8 WMMA/k-step),
//   async global->LDS double buffering.
//   mode 0: N=3072 -> scatter qb/kb/vb per-head [bh][l][d], q*SCALE
//   mode 1: N=1024 -> out32[m*1024+n] = acc + bo[n]
// ---------------------------------------------------------------------------
#define GBM 128
#define GBN 128
#define GBK 32
#define LDT 40   // row stride in halves (32 + 8 pad, 16B aligned)

__global__ __launch_bounds__(256) void wmma_gemm_kernel(
    const _Float16* __restrict__ A, const _Float16* __restrict__ W, int mode,
    const float* __restrict__ bqk, const float* __restrict__ bv,
    _Float16* __restrict__ qb, _Float16* __restrict__ kb, _Float16* __restrict__ vb,
    const float* __restrict__ bo, float* __restrict__ out32) {
  __shared__ _Float16 As[2][GBM * LDT];
  __shared__ _Float16 Bs[2][GBN * LDT];

  const int tid = threadIdx.x;
  const int lane = tid & 31, wid = tid >> 5;
  const int hl = lane >> 4, ln = lane & 15;
  const int m0 = blockIdx.y * GBM;
  const int n0 = blockIdx.x * GBN;
  const int wm = (wid & 3) * 32;      // 4 row groups x 32
  const int wn = (wid >> 2) * 64;     // 2 col groups x 64
  const int K = HIDDEN;
  const int nK = K / GBK;

  auto loadA = [&](int buf, int kt) {
    const int kb0 = kt * GBK;
#pragma unroll
    for (int it = 0; it < 2; ++it) {               // 512 chunks / 256 thr
      const int c = it * 256 + tid;
      const int r = c >> 2, o2 = (c & 3) * 8;
      copy16_g2l(&As[buf][r * LDT + o2], &A[(size_t)(m0 + r) * K + kb0 + o2]);
    }
  };
  auto loadB = [&](int buf, int kt) {
    const int kb0 = kt * GBK;
#pragma unroll
    for (int it = 0; it < 2; ++it) {
      const int c = it * 256 + tid;
      const int r = c >> 2, o2 = (c & 3) * 8;
      copy16_g2l(&Bs[buf][r * LDT + o2], &W[(size_t)(n0 + r) * K + kb0 + o2]);
    }
  };

  v8f acc[2][4] = {};
  loadA(0, 0);
  loadB(0, 0);
  async_wait0();
  __syncthreads();

  for (int kt = 0; kt < nK; ++kt) {
    const int cur = kt & 1, nxt = cur ^ 1;
    if (kt + 1 < nK) { loadA(nxt, kt + 1); loadB(nxt, kt + 1); }
    if (kt + 2 < nK)  // keep HBM stream ahead: global_prefetch_b8
      __builtin_prefetch(&A[(size_t)(m0 + (tid >> 1)) * K + (kt + 2) * GBK + (tid & 1) * 16], 0, 1);

    const _Float16* Ac = As[cur];
    const _Float16* Bc = Bs[cur];
    v16h a0 = load_a_frag(Ac, LDT, wm, 0);
    v16h a1 = load_a_frag(Ac, LDT, wm + 16, 0);
#pragma unroll
    for (int fj = 0; fj < 4; ++fj) {
      v16h b = load_b_frag(Bc, LDT, wn + fj * 16, 0);
      acc[0][fj] = WMMAF16(a0, b, acc[0][fj]);
      acc[1][fj] = WMMAF16(a1, b, acc[1][fj]);
    }
    async_wait0();
    __syncthreads();
  }

  for (int fi = 0; fi < 2; ++fi) {
    for (int fj = 0; fj < 4; ++fj) {
      const int n = n0 + wn + fj * 16 + ln;
      float bias;
      if (mode == 0) bias = (n < 2048) ? bqk[n] : bv[n - 2048];
      else           bias = bo[n];
      for (int j = 0; j < 8; ++j) {
        const int m = m0 + wm + fi * 16 + hl * 8 + j;
        const float val = acc[fi][fj][j] + bias;
        if (mode == 0) {
          const int lrow = m >> 4, bb = m & 15;          // m = l*B + b
          if (n < 1024) {
            const int hh = n >> 6, d = n & 63;
            qb[(((size_t)(bb * HEADS + hh) * SEQ + lrow) << 6) + d] = (_Float16)(val * SCALE);
          } else if (n < 2048) {
            const int n2 = n - 1024, hh = n2 >> 6, d = n2 & 63;
            kb[(((size_t)(bb * HEADS + hh) * SEQ + lrow) << 6) + d] = (_Float16)val;
          } else {
            const int n2 = n - 2048, hh = n2 >> 6, d = n2 & 63;
            vb[(((size_t)(bb * HEADS + hh) * SEQ + lrow) << 6) + d] = (_Float16)val;
          }
        } else {
          out32[(size_t)m * HIDDEN + n] = val;
        }
      }
    }
  }
}

// ---------------------------------------------------------------------------
// Kernel 5 (WMMA): positional score tables per (b,h):
//   QP[bh][l][buck] = q[bh][l] . kpos[h][buck]   (scale already in q)
//   PK[bh][l][buck] = k[bh][l] . qpos[h][buck]   (scale already in qpos)
// grid (SEQ/128, 256); wave wid handles 16 rows, frags cover 64 buckets.
// ---------------------------------------------------------------------------
__global__ __launch_bounds__(256) void pos_scores_kernel(const _Float16* __restrict__ qbuf,
                                                         const _Float16* __restrict__ kbuf,
                                                         const _Float16* __restrict__ qpos,
                                                         const _Float16* __restrict__ kpos,
                                                         _Float16* __restrict__ QP,
                                                         _Float16* __restrict__ PK) {
  __shared__ _Float16 Qt[128 * 72], Kt[128 * 72];
  __shared__ _Float16 KP[64 * 72], QPp[64 * 72];
  const int tid = threadIdx.x;
  const int lane = tid & 31, wid = tid >> 5;
  const int hl = lane >> 4, ln = lane & 15;
  const int bh = blockIdx.y, h = bh & 15;
  const int l0 = blockIdx.x * 128;
  const size_t rbase = ((size_t)bh * SEQ + l0) * 64;

#pragma unroll
  for (int it = 0; it < 4; ++it) {                 // 1024 chunks
    const int c = it * 256 + tid;
    const int r = c >> 3, o2 = (c & 7) * 8;
    copy16_g2l(&Qt[r * 72 + o2], &qbuf[rbase + r * 64 + o2]);
    copy16_g2l(&Kt[r * 72 + o2], &kbuf[rbase + r * 64 + o2]);
  }
#pragma unroll
  for (int it = 0; it < 2; ++it) {                 // 512 chunks
    const int c = it * 256 + tid;
    const int r = c >> 3, o2 = (c & 7) * 8;
    copy16_g2l(&KP[r * 72 + o2],  &kpos[(size_t)(h * 64 + r) * 64 + o2]);
    copy16_g2l(&QPp[r * 72 + o2], &qpos[(size_t)(h * 64 + r) * 64 + o2]);
  }
  async_wait0();
  __syncthreads();

  const int wm = wid * 16;
  v16h qa0 = load_a_frag(Qt, 72, wm, 0);
  v16h qa1 = load_a_frag(Qt, 72, wm, 32);
  v16h ka0 = load_a_frag(Kt, 72, wm, 0);
  v16h ka1 = load_a_frag(Kt, 72, wm, 32);
  v8f accQ[4], accP[4];
#pragma unroll
  for (int fj = 0; fj < 4; ++fj) {
    v8f z = {};
    v16h b0 = load_b_frag(KP, 72, fj * 16, 0);
    v16h b1 = load_b_frag(KP, 72, fj * 16, 32);
    v8f t = WMMAF16(qa0, b0, z);
    accQ[fj] = WMMAF16(qa1, b1, t);
    v16h c0 = load_b_frag(QPp, 72, fj * 16, 0);
    v16h c1 = load_b_frag(QPp, 72, fj * 16, 32);
    v8f u = WMMAF16(ka0, c0, z);
    accP[fj] = WMMAF16(ka1, c1, u);
  }
#pragma unroll
  for (int fj = 0; fj < 4; ++fj) {
    const int bucket = fj * 16 + ln;
    for (int j = 0; j < 8; ++j) {
      const int row = wm + hl * 8 + j;
      QP[rbase + (size_t)row * 64 + bucket] = (_Float16)accQ[fj][j];
      PK[rbase + (size_t)row * 64 + bucket] = (_Float16)accP[fj][j];
    }
  }
}

// ---------------------------------------------------------------------------
// Kernel 6: flash-style attention. grid=(8 q-tiles, 256 bh), 256 threads.
// ---------------------------------------------------------------------------
__global__ __launch_bounds__(256) void attn_kernel(const _Float16* __restrict__ qbuf,
                                                   const _Float16* __restrict__ kbuf,
                                                   const _Float16* __restrict__ vbuf,
                                                   const _Float16* __restrict__ QP,
                                                   const _Float16* __restrict__ PK,
                                                   const unsigned char* __restrict__ mask,
                                                   _Float16* __restrict__ ctx) {
  const int tid = threadIdx.x;
  const int lane = tid & 31, wid = tid >> 5;
  const int hl = lane >> 4, ln = lane & 15;
  const int qt = blockIdx.x;            // q tile (64 rows)
  const int bh = blockIdx.y;            // b*HEADS + h
  const int b = bh >> 4, h = bh & 15;
  const int wm = (wid & 3) * 16;        // 16 q rows per wave
  const int wn = (wid >> 2) * 32;       // 32 cols per wave

  __shared__ _Float16 Qs[64 * 72], Ks[64 * 72], Vs[64 * 72], Ps[64 * 72];
  __shared__ _Float16 QPs[64 * 72], PKs[64 * 72];
  __shared__ short rb[1024];
  __shared__ unsigned char mk[512];
  __shared__ float m_s[64], l_s[64], cf_s[64], mx_s[64];
  __shared__ float red_max[64][2], red_sum[64][2];

  for (int i = tid; i < 1023; i += 256) rb[i] = (short)bucket_of(i - 511);
  for (int i = tid; i < 512; i += 256) mk[i] = mask[b * SEQ + i];
  const size_t qbase = ((size_t)bh * SEQ + qt * 64) * 64;
#pragma unroll
  for (int it = 0; it < 2; ++it) {
    const int c = it * 256 + tid;
    const int r = c >> 3, o2 = (c & 7) * 8;
    copy16_g2l(&Qs[r * 72 + o2],  &qbuf[qbase + r * 64 + o2]);
    copy16_g2l(&QPs[r * 72 + o2], &QP[qbase + r * 64 + o2]);
  }
  if (tid < 64) { m_s[tid] = -3.0e38f; l_s[tid] = 0.f; }
  v8f cacc[2] = {};
  async_wait0();
  __syncthreads();

  for (int kt = 0; kt < 8; ++kt) {
    const size_t kbase = ((size_t)bh * SEQ + kt * 64) * 64;
#pragma unroll
    for (int it = 0; it < 2; ++it) {
      const int c = it * 256 + tid;
      const int r = c >> 3, o2 = (c & 7) * 8;
      copy16_g2l(&Ks[r * 72 + o2],  &kbuf[kbase + r * 64 + o2]);
      copy16_g2l(&PKs[r * 72 + o2], &PK[kbase + r * 64 + o2]);
    }
#pragma unroll
    for (int it = 0; it < 2; ++it) {               // V transposed: Vs[d][k]
      const int c = it * 256 + tid;
      const int kk = c >> 3, o2 = (c & 7) * 8;
      union { uint4 u; _Float16 e[8]; } t;
      t.u = *(const uint4*)&vbuf[kbase + kk * 64 + o2];
#pragma unroll
      for (int x = 0; x < 8; ++x) Vs[(o2 + x) * 72 + kk] = t.e[x];
    }
    async_wait0();
    __syncthreads();

    // --- S = Q K^T (K=64: two 16x16x32 WMMAs per frag) ---
    v8f z = {};
    v16h a0 = load_a_frag(Qs, 72, wm, 0);
    v16h a1 = load_a_frag(Qs, 72, wm, 32);
    v8f sacc[2];
#pragma unroll
    for (int fj = 0; fj < 2; ++fj) {
      v16h b0 = load_b_frag(Ks, 72, wn + fj * 16, 0);
      v16h b1 = load_b_frag(Ks, 72, wn + fj * 16, 32);
      v8f s = WMMAF16(a0, b0, z);
      sacc[fj] = WMMAF16(a1, b1, s);
    }

    // --- add disentangled terms + mask, row max ---
    float vmax[8];
    for (int j = 0; j < 8; ++j) vmax[j] = -3.0e38f;
    for (int fj = 0; fj < 2; ++fj) {
      for (int j = 0; j < 8; ++j) {
        const int qr = wm + hl * 8 + j;
        const int kr = wn + fj * 16 + ln;
        const int qg = qt * 64 + qr, kg = kt * 64 + kr;
        const int bi = rb[qg - kg + 511];
        float s = sacc[fj][j] + (float)QPs[qr * 72 + bi] + (float)PKs[kr * 72 + bi];
        if (mk[kg]) s = -1.0e30f;
        sacc[fj][j] = s;
        vmax[j] = fmaxf(vmax[j], s);
      }
    }
    for (int j = 0; j < 8; ++j)
      for (int o = 1; o < 16; o <<= 1)
        vmax[j] = fmaxf(vmax[j], __shfl_xor(vmax[j], o, 32));
    if (ln == 0)
      for (int j = 0; j < 8; ++j) red_max[wm + hl * 8 + j][wid >> 2] = vmax[j];
    __syncthreads();

    if (tid < 64) {
      const float nm = fmaxf(m_s[tid], fmaxf(red_max[tid][0], red_max[tid][1]));
      cf_s[tid] = __expf(m_s[tid] - nm);
      mx_s[tid] = nm;
      m_s[tid] = nm;
    }
    __syncthreads();

    // --- P = exp(S - m) (masked -> 0), stage to LDS, row sums ---
    float vsum[8];
    for (int j = 0; j < 8; ++j) vsum[j] = 0.f;
    for (int fj = 0; fj < 2; ++fj) {
      for (int j = 0; j < 8; ++j) {
        const int qr = wm + hl * 8 + j;
        const int kr = wn + fj * 16 + ln;
        const int kg = kt * 64 + kr;
        const float p = mk[kg] ? 0.f : __expf(sacc[fj][j] - mx_s[qr]);
        Ps[qr * 72 + kr] = (_Float16)p;
        vsum[j] += p;
      }
    }
    for (int j = 0; j < 8; ++j)
      for (int o = 1; o < 16; o <<= 1)
        vsum[j] += __shfl_xor(vsum[j], o, 32);
    if (ln == 0)
      for (int j = 0; j < 8; ++j) red_sum[wm + hl * 8 + j][wid >> 2] = vsum[j];
    __syncthreads();

    if (tid < 64) l_s[tid] = l_s[tid] * cf_s[tid] + red_sum[tid][0] + red_sum[tid][1];

    // --- rescale ctx, ctx += P @ V ---
    for (int fj = 0; fj < 2; ++fj)
      for (int j = 0; j < 8; ++j)
        cacc[fj][j] *= cf_s[wm + hl * 8 + j];
    v16h p0 = load_a_frag(Ps, 72, wm, 0);
    v16h p1 = load_a_frag(Ps, 72, wm, 32);
#pragma unroll
    for (int fj = 0; fj < 2; ++fj) {
      v16h v0 = load_b_frag(Vs, 72, wn + fj * 16, 0);
      v16h v1 = load_b_frag(Vs, 72, wn + fj * 16, 32);
      cacc[fj] = WMMAF16(p0, v0, cacc[fj]);
      cacc[fj] = WMMAF16(p1, v1, cacc[fj]);
    }
    __syncthreads();
  }

  // --- finalize: ctx / l, write (L,B,H*64) f16 ---
  for (int fj = 0; fj < 2; ++fj) {
    for (int j = 0; j < 8; ++j) {
      const int qr = wm + hl * 8 + j;
      const int qg = qt * 64 + qr;
      const float l = l_s[qr];
      const float inv = l > 0.f ? 1.f / l : 0.f;
      const int d = wn + fj * 16 + ln;
      const size_t m = (size_t)qg * BATCH + b;
      ctx[m * HIDDEN + h * 64 + d] = (_Float16)(cacc[fj][j] * inv);
    }
  }
}

// ---------------------------------------------------------------------------
// Kernel 7: output LayerNorm with affine -> d_out (fp32)
// ---------------------------------------------------------------------------
__global__ __launch_bounds__(256) void ln_out_kernel(const float* __restrict__ x,
                                                     const float* __restrict__ g,
                                                     const float* __restrict__ bb,
                                                     float* __restrict__ out) {
  __shared__ float ssum[8], ssq[8];
  const int tid = threadIdx.x;
  const size_t row = blockIdx.x;
  const float4* xr = (const float4*)(x + row * HIDDEN);
  float4 v = xr[tid];
  float s = v.x + v.y + v.z + v.w;
  float q = v.x * v.x + v.y * v.y + v.z * v.z + v.w * v.w;
  for (int o = 16; o >= 1; o >>= 1) { s += __shfl_xor(s, o, 32); q += __shfl_xor(q, o, 32); }
  if ((tid & 31) == 0) { ssum[tid >> 5] = s; ssq[tid >> 5] = q; }
  __syncthreads();
  float ts = 0.f, tq = 0.f;
  for (int i = 0; i < 8; ++i) { ts += ssum[i]; tq += ssq[i]; }
  const float mean = ts * (1.0f / HIDDEN);
  const float var  = tq * (1.0f / HIDDEN) - mean * mean;
  const float rstd = rsqrtf(var + 1e-7f);
  const float4 gg = ((const float4*)g)[tid];
  const float4 bbv = ((const float4*)bb)[tid];
  float4 o;
  o.x = (v.x - mean) * rstd * gg.x + bbv.x;
  o.y = (v.y - mean) * rstd * gg.y + bbv.y;
  o.z = (v.z - mean) * rstd * gg.z + bbv.z;
  o.w = (v.w - mean) * rstd * gg.w + bbv.w;
  ((float4*)(out + row * HIDDEN))[tid] = o;
}

// ---------------------------------------------------------------------------
// Host orchestration
// ---------------------------------------------------------------------------
extern "C" void kernel_launch(void* const* d_in, const int* in_sizes, int n_in,
                              void* d_out, int out_size, void* d_ws, size_t ws_size,
                              hipStream_t stream) {
  const float* hidden = (const float*)d_in[0];
  const unsigned char* mask = (const unsigned char*)d_in[1];
  const float* rel = (const float*)d_in[2];
  const float* wqk = (const float*)d_in[3];
  const float* bqk = (const float*)d_in[4];
  const float* wv  = (const float*)d_in[5];
  const float* bv  = (const float*)d_in[6];
  const float* wo  = (const float*)d_in[7];
  const float* bo  = (const float*)d_in[8];
  const float* lng = (const float*)d_in[9];
  const float* lnb = (const float*)d_in[10];
  float* out = (float*)d_out;

  char* ws = (char*)d_ws;
  size_t off = 0;
  auto alloc = [&](size_t bytes) { size_t o = off; off += (bytes + 255) & ~(size_t)255; return o; };
  const size_t o_h16  = alloc((size_t)NROWS * HIDDEN * 2);
  const size_t o_wcat = alloc((size_t)3072 * 1024 * 2);
  const size_t o_wo16 = alloc((size_t)1024 * 1024 * 2);
  const size_t o_qpos = alloc((size_t)16 * 64 * 64 * 2);
  const size_t o_kpos = alloc((size_t)16 * 64 * 64 * 2);
  const size_t o_qb   = alloc((size_t)256 * SEQ * 64 * 2);
  const size_t o_kb   = alloc((size_t)256 * SEQ * 64 * 2);
  const size_t o_vb   = alloc((size_t)256 * SEQ * 64 * 2);
  const size_t o_QP   = alloc((size_t)256 * SEQ * 64 * 2);
  const size_t o_PK   = alloc((size_t)256 * SEQ * 64 * 2);
  const size_t o_ctx  = alloc((size_t)NROWS * HIDDEN * 2);
  const size_t o_o32  = alloc((size_t)NROWS * HIDDEN * 4);

  _Float16* h16  = (_Float16*)(ws + o_h16);
  _Float16* wcat = (_Float16*)(ws + o_wcat);
  _Float16* wo16 = (_Float16*)(ws + o_wo16);
  _Float16* qpos = (_Float16*)(ws + o_qpos);
  _Float16* kpos = (_Float16*)(ws + o_kpos);
  _Float16* qb   = (_Float16*)(ws + o_qb);
  _Float16* kb   = (_Float16*)(ws + o_kb);
  _Float16* vb   = (_Float16*)(ws + o_vb);
  _Float16* QPb  = (_Float16*)(ws + o_QP);
  _Float16* PKb  = (_Float16*)(ws + o_PK);
  _Float16* ctx  = (_Float16*)(ws + o_ctx);
  float*    o32  = (float*)(ws + o_o32);

  ln_in_kernel<<<NROWS, 256, 0, stream>>>(hidden, h16);
  cvt_w_kernel<<<(3072 * 1024 + 255) / 256, 256, 0, stream>>>(wqk, wv, wo, wcat, wo16);
  pos_proj_kernel<<<2048, 64, 0, stream>>>(rel, wqk, bqk, qpos, kpos);
  wmma_gemm_kernel<<<dim3(3072 / GBN, NROWS / GBM), 256, 0, stream>>>(
      h16, wcat, 0, bqk, bv, qb, kb, vb, nullptr, nullptr);
  pos_scores_kernel<<<dim3(SEQ / 128, 256), 256, 0, stream>>>(qb, kb, qpos, kpos, QPb, PKb);
  attn_kernel<<<dim3(SEQ / 64, 256), 256, 0, stream>>>(qb, kb, vb, QPb, PKb, mask, ctx);
  wmma_gemm_kernel<<<dim3(1024 / GBN, NROWS / GBM), 256, 0, stream>>>(
      ctx, wo16, 1, nullptr, nullptr, nullptr, nullptr, nullptr, bo, o32);
  ln_out_kernel<<<NROWS, 256, 0, stream>>>(o32, lng, lnb, out);
}
